// ChessNet_23562190586190
// MI455X (gfx1250) — matrix operand order repaired and checked by reading, not compiled, not run on hardware
//
#include <hip/hip_runtime.h>

// ---------------------------------------------------------------------------
// ChessNet-style transformer forward for MI455X (gfx1250), wave32 + WMMA bf16.
// GEMMs use v_wmma_f32_16x16x32_bf16, 2x2 register blocking per wave,
// double-buffered LDS staging (one barrier per K-step), and — when the
// toolchain provides the builtins — GLOBAL_LOAD_ASYNC_TO_LDS_B128 staging
// with deferred s_wait_asynccnt (ASYNCcnt overlapped with the WMMAs).
// ---------------------------------------------------------------------------

#define BATCH  4
#define SEQ    2048
#define KD     512
#define HEADS  8
#define HDIM   64
#define LAYERS 8
#define VOCAB  71
#define MTOK   (BATCH * SEQ)   // 8192 tokens
#define FF     2048

typedef __attribute__((ext_vector_type(16))) __bf16        v16bf;
typedef __attribute__((ext_vector_type(8)))  float         v8f;
typedef __attribute__((ext_vector_type(4)))  unsigned int  u32x4;
typedef __attribute__((ext_vector_type(4)))  int           v4i;

#if defined(__AMDGCN__) && \
    __has_builtin(__builtin_amdgcn_global_load_async_to_lds_b128) && \
    __has_builtin(__builtin_amdgcn_s_wait_asynccnt)
#define USE_ASYNC_LDS 1
typedef __attribute__((address_space(1))) v4i global_v4i;
typedef __attribute__((address_space(3))) v4i lds_v4i;
#else
#define USE_ASYNC_LDS 0
#endif

union BF16x16 {
  v16bf        v;
  unsigned int d[8];
};

__device__ __forceinline__ unsigned short f2bfu(float f) {
  union { float f; unsigned int u; } b;
  b.f = f;
  unsigned int r = b.u + 0x7fffu + ((b.u >> 16) & 1u);   // round-to-nearest-even
  return (unsigned short)(r >> 16);
}

__device__ __forceinline__ void async_wait_all() {
#if USE_ASYNC_LDS
  __builtin_amdgcn_s_wait_asynccnt(0);
#endif
}

// ---------------------------------------------------------------------------
// Embedding gather: r[t, c] = emb[x[t], c]
// ---------------------------------------------------------------------------
__global__ __launch_bounds__(256) void k_embed(const int* __restrict__ x,
                                               const float* __restrict__ emb,
                                               float* __restrict__ r) {
  int idx = blockIdx.x * 256 + threadIdx.x;   // MTOK*KD threads exactly
  int t = idx >> 9;                            // /512
  int c = idx & 511;
  r[idx] = emb[(size_t)x[t] * KD + c];
}

// ---------------------------------------------------------------------------
// LayerNorm over KD=512, one block per row; writes fp32 and bf16 copies.
// ---------------------------------------------------------------------------
__global__ __launch_bounds__(256) void k_layernorm(const float* __restrict__ in,
                                                   const float* __restrict__ g,
                                                   const float* __restrict__ b,
                                                   float* __restrict__ outF,
                                                   unsigned short* __restrict__ outB) {
  __shared__ float red[256];
  int row = blockIdx.x, tid = threadIdx.x;
  const float* p = in + (size_t)row * KD;
  float x0 = p[tid], x1 = p[tid + 256];

  red[tid] = x0 + x1;
  __syncthreads();
  for (int s = 128; s > 0; s >>= 1) { if (tid < s) red[tid] += red[tid + s]; __syncthreads(); }
  float mu = red[0] * (1.0f / KD);
  __syncthreads();

  float d0 = x0 - mu, d1 = x1 - mu;
  red[tid] = d0 * d0 + d1 * d1;
  __syncthreads();
  for (int s = 128; s > 0; s >>= 1) { if (tid < s) red[tid] += red[tid + s]; __syncthreads(); }
  float inv = rsqrtf(red[0] * (1.0f / KD) + 1e-5f);

  float y0 = d0 * inv * g[tid]       + b[tid];
  float y1 = d1 * inv * g[tid + 256] + b[tid + 256];
  size_t o = (size_t)row * KD;
  outF[o + tid]       = y0;
  outF[o + tid + 256] = y1;
  outB[o + tid]       = f2bfu(y0);
  outB[o + tid + 256] = f2bfu(y1);
}

// ---------------------------------------------------------------------------
// RoPE (interleaved pairs, matches reference) + pack to bf16.
// ---------------------------------------------------------------------------
__global__ __launch_bounds__(256) void k_rope(const float* __restrict__ in,
                                              unsigned short* __restrict__ out) {
  int idx = blockIdx.x * 256 + threadIdx.x;   // MTOK * 256 pairs
  int t  = idx >> 8;
  int pp = idx & 255;
  int hh = pp >> 5;
  int j  = pp & 31;
  int n  = t & (SEQ - 1);
  float ang = (float)n * expf((float)j * (-9.210340371976184f / 32.0f));
  float s, c;
  sincosf(ang, &s, &c);
  size_t base = (size_t)t * KD + hh * HDIM + 2 * j;
  float x1 = in[base], x2 = in[base + 1];
  out[base]     = f2bfu( x1 * c + x2 * s);
  out[base + 1] = f2bfu(-x1 * s + x2 * c);
}

// ---------------------------------------------------------------------------
// Elementwise fp32 -> bf16
// ---------------------------------------------------------------------------
__global__ __launch_bounds__(256) void k_cvt_bf16(const float* __restrict__ in,
                                                  unsigned short* __restrict__ out,
                                                  int n) {
  int i = blockIdx.x * 256 + threadIdx.x;
  if (i < n) out[i] = f2bfu(in[i]);
}

// ---------------------------------------------------------------------------
// Causal row softmax: S fp32 [SEQ,SEQ] -> P bf16 [SEQ,SEQ]; cols > row get 0.
// ---------------------------------------------------------------------------
__global__ __launch_bounds__(256) void k_softmax(const float* __restrict__ S,
                                                 unsigned short* __restrict__ P) {
  __shared__ float red[256];
  int row = blockIdx.x, tid = threadIdx.x;
  const float* sp = S + (size_t)row * SEQ;
  float v[8];
  float m = -3.0e38f;
#pragma unroll
  for (int i = 0; i < 8; ++i) {
    int c = tid + i * 256;
    float t = (c <= row) ? sp[c] : -3.0e38f;
    v[i] = t;
    m = fmaxf(m, t);
  }
  red[tid] = m;
  __syncthreads();
  for (int s = 128; s > 0; s >>= 1) { if (tid < s) red[tid] = fmaxf(red[tid], red[tid + s]); __syncthreads(); }
  float mx = red[0];
  __syncthreads();
  float sum = 0.0f;
#pragma unroll
  for (int i = 0; i < 8; ++i) {
    int c = tid + i * 256;
    float e = (c <= row) ? expf(v[i] - mx) : 0.0f;
    v[i] = e;
    sum += e;
  }
  red[tid] = sum;
  __syncthreads();
  for (int s = 128; s > 0; s >>= 1) { if (tid < s) red[tid] += red[tid + s]; __syncthreads(); }
  float inv = 1.0f / red[0];
#pragma unroll
  for (int i = 0; i < 8; ++i) {
    int c = tid + i * 256;
    P[(size_t)row * SEQ + c] = f2bfu(v[i] * inv);
  }
}

// ---------------------------------------------------------------------------
// bf16 WMMA GEMM: C[M,N] = A[M,K] * B[K,N], fp32 accumulate.
//   block = 256 threads = 8 waves (4M x 2N); wave tile 32x32 = 2x2 WMMA tiles;
//   block tile 128x64; K-step 32; double-buffered LDS, 1 barrier per step.
//   A tile and transposed-B tile are staged with async-to-LDS copies when the
//   toolchain exposes them (ASYNCcnt waited only after the WMMAs).
//   bFlags: bit0 = B is bf16 (else fp32, converted while staging)
//           bit1 = B stored transposed as [N][K] (row n contiguous in K)
//   epilogue: v = acc*scale (+bias[col]) (gelu) (+resid[row,col]) -> outF/outB
//   Requirements (all satisfied by every call site): M % 128 == 0,
//   K % 32 == 0, bf16-B paths need N % 64 == 0; fp32-B path zero-pads N.
// ---------------------------------------------------------------------------
#define BM 128
#define BN 64
#define KC 32

__global__ __launch_bounds__(256)
void k_gemm(const unsigned short* __restrict__ A, int lda,
            const void* __restrict__ B, int ldb, int bFlags,
            const float* __restrict__ bias,
            const float* __restrict__ resid, int ldr,
            float* __restrict__ outF, int ldo,
            unsigned short* __restrict__ outB, int ldob,
            int N, int Kdim, float scale, int doGelu) {
  __shared__ __align__(16) unsigned short sA[2][BM][KC];   // 16 KB
  __shared__ __align__(16) unsigned short sB[2][BN][KC];   //  8 KB, B^T: [n][k]

  const int tid  = threadIdx.x;
  const int lane = tid & 31;
  const int wave = tid >> 5;
  const int wm   = wave >> 1;       // 0..3  (M sub-block, 32 rows each)
  const int wn   = wave & 1;        // 0..1  (N sub-block, 32 cols each)
  const int hl   = lane >> 4;       // lane half (ISA fragment layouts)
  const int l16  = lane & 15;
  const int m0   = blockIdx.x * BM;
  const int n0   = blockIdx.y * BN;

  const u32x4* A128 = (const u32x4*)A;

  // stage one K-slab (KC columns) into LDS buffer `buf`
  auto stage = [&](int buf, int kb) {
    // ---- A tile: 128x32 bf16 = 512 u32x4, 2 per thread ----
#pragma unroll
    for (int i = 0; i < 2; ++i) {
      int e4 = tid + i * 256;                 // 0..511
      int rr = e4 >> 2;                       // tile row
      int c4 = e4 & 3;                        // 16B column chunk
      unsigned int off = (((unsigned int)(m0 + rr) * lda + kb) >> 3) + c4;
#if USE_ASYNC_LDS
      __builtin_amdgcn_global_load_async_to_lds_b128(
          (global_v4i*)(A128 + off), (lds_v4i*)&((u32x4*)sA[buf])[e4], 0, 0);
#else
      ((u32x4*)sA[buf])[e4] = A128[off];
#endif
    }
    // ---- B tile as [n][k] ----
    if (bFlags & 2) {                          // bf16, already [N][K]
      const u32x4* B128 = (const u32x4*)B;
      int e4 = tid;                            // 256 u32x4, 1 per thread
      int nn = e4 >> 2;
      int c4 = e4 & 3;
      unsigned int off = (((unsigned int)(n0 + nn) * ldb + kb) >> 3) + c4;
#if USE_ASYNC_LDS
      __builtin_amdgcn_global_load_async_to_lds_b128(
          (global_v4i*)(B128 + off), (lds_v4i*)&((u32x4*)sB[buf])[e4], 0, 0);
#else
      ((u32x4*)sB[buf])[e4] = B128[off];
#endif
    } else if (bFlags & 1) {                   // bf16, [K][N] (strided gather)
      const unsigned short* B16 = (const unsigned short*)B;
#pragma unroll
      for (int i = 0; i < 8; ++i) {
        int e  = tid + i * 256;
        int kk = e >> 6;
        int nn = e & 63;
        sB[buf][nn][kk] = B16[(size_t)(kb + kk) * ldb + (n0 + nn)];
      }
    } else {                                   // fp32 weights, convert + pad N
      const float* Bf = (const float*)B;
#pragma unroll
      for (int i = 0; i < 8; ++i) {
        int e  = tid + i * 256;
        int kk = e >> 6;
        int nn = e & 63;
        float v = (n0 + nn < N) ? Bf[(size_t)(kb + kk) * ldb + (n0 + nn)] : 0.0f;
        sB[buf][nn][kk] = f2bfu(v);
      }
      // prefetch next weight slab into cache (global_prefetch_b8)
      if (tid < KC && kb + KC < Kdim)
        __builtin_prefetch(&Bf[(size_t)(kb + KC + tid) * ldb + n0], 0, 0);
    }
  };

  v8f acc[2][2];
  acc[0][0] = v8f{}; acc[0][1] = v8f{}; acc[1][0] = v8f{}; acc[1][1] = v8f{};

  const int nK = Kdim / KC;
  stage(0, 0);
  async_wait_all();
  __syncthreads();

  for (int ki = 0; ki < nK; ++ki) {
    const int cur = ki & 1;
    if (ki + 1 < nK) stage(cur ^ 1, (ki + 1) * KC);

    // ---- fragments (ISA 7.12.2 16-bit layouts) ----
    BF16x16 fa[2], fb[2];
#pragma unroll
    for (int tm = 0; tm < 2; ++tm) {
      const unsigned short* ap = sA[cur][wm * 32 + tm * 16 + l16];
#pragma unroll
      for (int p = 0; p < 8; ++p) {
        // A: lane holds row l16; K pairs [hl*8 .. +7] and [16+hl*8 .. +7]
        int kk2 = (p < 4) ? (hl * 8 + 2 * p) : (16 + hl * 8 + 2 * (p - 4));
        fa[tm].d[p] = *(const unsigned int*)&ap[kk2];
      }
    }
#pragma unroll
    for (int tn = 0; tn < 2; ++tn) {
      const unsigned short* bp = sB[cur][wn * 32 + tn * 16 + l16];
#pragma unroll
      for (int p = 0; p < 8; ++p) {
        // B: lane holds col l16; K pairs [hl*8 .. +7] and [16+hl*8 .. +7]
        fb[tn].d[p] = *(const unsigned int*)&bp[(p < 4) ? (hl * 8 + 2 * p) : (16 + hl * 8 + 2 * (p - 4))];
      }
    }
    // 4 independent WMMAs (distinct accumulators -> co-executable)
#pragma unroll
    for (int tm = 0; tm < 2; ++tm)
#pragma unroll
      for (int tn = 0; tn < 2; ++tn)
        acc[tm][tn] = __builtin_amdgcn_wmma_f32_16x16x32_bf16(
            false, fa[tm].v, false, fb[tn].v, (short)0, acc[tm][tn], false, false);

    async_wait_all();     // async staging for next slab must land before...
    __syncthreads();      // ...the buffer is published to all waves
  }

  // ---- epilogue: D layout row = 8*hl + i, col = l16 ----
#pragma unroll
  for (int tn = 0; tn < 2; ++tn) {
    int col = n0 + wn * 32 + tn * 16 + l16;
    if (col < N) {
      float bv = bias ? bias[col] : 0.0f;
#pragma unroll
      for (int tm = 0; tm < 2; ++tm) {
#pragma unroll
        for (int i = 0; i < 8; ++i) {
          int row = m0 + wm * 32 + tm * 16 + hl * 8 + i;
          float v = acc[tm][tn][i] * scale + bv;
          if (doGelu) v = 0.5f * v * (1.0f + erff(v * 0.70710678118654752f));
          if (resid)  v += resid[(size_t)row * ldr + col];
          if (outF)   outF[(size_t)row * ldo + col] = v;
          if (outB)   outB[(size_t)row * ldob + col] = f2bfu(v);
        }
      }
    }
  }
}

// ---------------------------------------------------------------------------
// Orchestration. Workspace layout (136 MiB total):
//   r, xn, tmp : fp32 [MTOK,512]      (16 MiB each)
//   S          : fp32 [SEQ,SEQ]       (16 MiB)
//   qb,kb,vb   : bf16 [MTOK,512]      ( 8 MiB each)
//   xnb        : bf16 [MTOK,512]      ( 8 MiB)
//   P          : bf16 [SEQ,SEQ]      ( 8 MiB)
//   hbuf       : bf16 [MTOK,2048]     (32 MiB)
// ---------------------------------------------------------------------------
extern "C" void kernel_launch(void* const* d_in, const int* in_sizes, int n_in,
                              void* d_out, int out_size, void* d_ws, size_t ws_size,
                              hipStream_t stream) {
  (void)in_sizes; (void)n_in; (void)out_size; (void)ws_size;

  const int*   x    = (const int*)  d_in[0];
  /* d_in[1] = mask: causality applied analytically in k_softmax */
  const float* emb  = (const float*)d_in[2];
  const float* Wq   = (const float*)d_in[3];
  const float* Wk   = (const float*)d_in[4];
  const float* Wv   = (const float*)d_in[5];
  const float* Wo   = (const float*)d_in[6];
  const float* bo   = (const float*)d_in[7];
  const float* ln1g = (const float*)d_in[8];
  const float* ln1b = (const float*)d_in[9];
  const float* ln2g = (const float*)d_in[10];
  const float* ln2b = (const float*)d_in[11];
  const float* W1   = (const float*)d_in[12];
  const float* b1   = (const float*)d_in[13];
  const float* W2   = (const float*)d_in[14];
  const float* b2   = (const float*)d_in[15];
  const float* Wl4  = (const float*)d_in[16];
  const float* bl4  = (const float*)d_in[17];

  char* w = (char*)d_ws;
  float* r    = (float*)w;  w += (size_t)MTOK * KD * 4;
  float* xn   = (float*)w;  w += (size_t)MTOK * KD * 4;
  float* tmp  = (float*)w;  w += (size_t)MTOK * KD * 4;
  float* S    = (float*)w;  w += (size_t)SEQ * SEQ * 4;
  unsigned short* qb   = (unsigned short*)w; w += (size_t)MTOK * KD * 2;
  unsigned short* kbuf = (unsigned short*)w; w += (size_t)MTOK * KD * 2;
  unsigned short* vb   = (unsigned short*)w; w += (size_t)MTOK * KD * 2;
  unsigned short* xnb  = (unsigned short*)w; w += (size_t)MTOK * KD * 2;
  unsigned short* P    = (unsigned short*)w; w += (size_t)SEQ * SEQ * 2;
  unsigned short* hbuf = (unsigned short*)w; w += (size_t)MTOK * FF * 2;

  const dim3 blk(256);
  const dim3 gProj(MTOK / BM, KD / BN);               //  64 x  8
  const dim3 gF1  (MTOK / BM, FF / BN);               //  64 x 32
  const dim3 gQK  (SEQ  / BM, SEQ / BN);              //  16 x 32
  const dim3 gPV  (SEQ  / BM, HDIM / BN);             //  16 x  1
  const dim3 gVoc (MTOK / BM, (VOCAB + BN - 1) / BN); //  64 x  2
  const int  nElem = MTOK * KD;

  k_embed<<<nElem / 256, blk, 0, stream>>>(x, emb, r);

  for (int l = 0; l < LAYERS; ++l) {
    const float* wq = Wq + (size_t)l * KD * KD;
    const float* wk = Wk + (size_t)l * KD * KD;
    const float* wv = Wv + (size_t)l * KD * KD;
    const float* wo = Wo + (size_t)l * KD * KD;
    const float* w1 = W1 + (size_t)l * KD * FF;
    const float* w2 = W2 + (size_t)l * FF * KD;

    // xn = LN1(r) (fp32 + bf16)
    k_layernorm<<<MTOK, blk, 0, stream>>>(r, ln1g + l * KD, ln1b + l * KD, xn, xnb);

    // Q = rope(xn @ Wq), K = rope(xn @ Wk), V = xn @ Wv   (all -> bf16)
    k_gemm<<<gProj, blk, 0, stream>>>(xnb, KD, wq, KD, 0, nullptr, nullptr, 0,
                                      tmp, KD, nullptr, 0, KD, KD, 1.0f, 0);
    k_rope<<<MTOK, blk, 0, stream>>>(tmp, qb);
    k_gemm<<<gProj, blk, 0, stream>>>(xnb, KD, wk, KD, 0, nullptr, nullptr, 0,
                                      tmp, KD, nullptr, 0, KD, KD, 1.0f, 0);
    k_rope<<<MTOK, blk, 0, stream>>>(tmp, kbuf);
    k_gemm<<<gProj, blk, 0, stream>>>(xnb, KD, wv, KD, 0, nullptr, nullptr, 0,
                                      tmp, KD, nullptr, 0, KD, KD, 1.0f, 0);
    k_cvt_bf16<<<nElem / 256, blk, 0, stream>>>(tmp, vb, nElem);

    // attention per (batch, head); o written into tmp (disjoint slices)
    for (int b = 0; b < BATCH; ++b) {
      for (int hh = 0; hh < HEADS; ++hh) {
        size_t off = (size_t)b * SEQ * KD + (size_t)hh * HDIM;
        // S = (Q K^T) / sqrt(64): B is K-matrix viewed as [N=kj][K=d]
        k_gemm<<<gQK, blk, 0, stream>>>(qb + off, KD, kbuf + off, KD, /*bf16|T*/3,
                                        nullptr, nullptr, 0, S, SEQ, nullptr, 0,
                                        SEQ, HDIM, 0.125f, 0);
        k_softmax<<<SEQ, blk, 0, stream>>>(S, P);
        // o = P @ V
        k_gemm<<<gPV, blk, 0, stream>>>(P, SEQ, vb + off, KD, /*bf16*/1,
                                        nullptr, nullptr, 0, tmp + off, KD,
                                        nullptr, 0, HDIM, SEQ, 1.0f, 0);
      }
    }

    // r2 = xn + (o @ Wo + bo)   (residual from the *normed* input, per ref)
    k_cvt_bf16<<<nElem / 256, blk, 0, stream>>>(tmp, qb, nElem);
    k_gemm<<<gProj, blk, 0, stream>>>(qb, KD, wo, KD, 0, bo + l * KD, xn, KD,
                                      r, KD, nullptr, 0, KD, KD, 1.0f, 0);

    // x2n = LN2(r2)  (reuse xn / xnb)
    k_layernorm<<<MTOK, blk, 0, stream>>>(r, ln2g + l * KD, ln2b + l * KD, xn, xnb);

    // h = gelu(x2n @ W1 + b1)  (bf16) ; r = x2n + (h @ W2 + b2)
    k_gemm<<<gF1, blk, 0, stream>>>(xnb, KD, w1, FF, 0, b1 + l * FF, nullptr, 0,
                                    nullptr, 0, hbuf, FF, FF, KD, 1.0f, 1);
    k_gemm<<<gProj, blk, 0, stream>>>(hbuf, FF, w2, KD, 0, b2 + l * KD, xn, KD,
                                      r, KD, nullptr, 0, KD, FF, 1.0f, 0);
  }

  // logits = r @ Wl4 + bl4  -> d_out [MTOK, 71]
  k_cvt_bf16<<<nElem / 256, blk, 0, stream>>>(r, xnb, nElem);
  k_gemm<<<gVoc, blk, 0, stream>>>(xnb, KD, Wl4, VOCAB, 0, bl4, nullptr, 0,
                                   (float*)d_out, VOCAB, nullptr, 0,
                                   VOCAB, KD, 1.0f, 0);
}
